// LuhnClassifier_52793738002710
// MI455X (gfx1250) — compile-verified, matrix operand order checked by reading, not run on hardware
//
#include <hip/hip_runtime.h>
#include <hip/hip_bf16.h>
#include <math.h>

typedef _Float16 v16h __attribute__((ext_vector_type(16)));
typedef _Float16 v8h  __attribute__((ext_vector_type(8)));
typedef float    v8f  __attribute__((ext_vector_type(8)));
// unaligned-capable 8 x f16 view (2-byte aligned loads)
typedef _Float16 v8h_u __attribute__((ext_vector_type(8))) __attribute__((aligned(2)));

// ---- WMMA f16 A/B per-lane element -> k mapping (16x32 A, 32x16 B) -------
// lanes 0-15: elems 0..7 -> k = 0..7,   elems 8..15 -> k = 16..23
// lanes16-31: elems 0..7 -> k = 8..15,  elems 8..15 -> k = 24..31
__device__ __forceinline__ int wmma_kof(int lane, int j) {
    int half = (lane >> 4) & 1;
    return (j < 8) ? (half * 8 + j) : (16 + half * 8 + (j - 8));
}

// ---- convert input images f32 -> f16 (with small tail pad of zeros) ------
__global__ __launch_bounds__(256) void cvt_f16(const float* __restrict__ src,
                                               _Float16* __restrict__ dst,
                                               int n, int total) {
    int i = blockIdx.x * 256 + threadIdx.x;
    if (i >= total) return;
    dst[i] = (i < n) ? (_Float16)src[i] : (_Float16)0;
}

// ---- Pack a KxN f32 weight matrix into WMMA-ready f16 B tiles ------------
// dst layout: [kt][tN][lane(32)][16 f16]  (one aligned v16h per lane per tile)
__global__ __launch_bounds__(256) void pack_b(const float* __restrict__ src,
                                              _Float16* __restrict__ dst,
                                              int K, int N, int tilesN, int total) {
    int e = blockIdx.x * 256 + threadIdx.x;
    if (e >= total) return;
    int j    = e & 15;
    int lane = (e >> 4) & 31;
    int tile = e >> 9;
    int tN   = tile % tilesN;
    int kt   = tile / tilesN;
    int k = kt * 32 + wmma_kof(lane, j);
    int n = tN * 16 + (lane & 15);
    float v = 0.0f;
    if (k < K && n < N) v = src[(size_t)k * N + n];
    dst[e] = (_Float16)v;
}

// ---- Pack cw1 with conv1 K-permutation: k' = kh*8 + kw (kw<5, kh<5) ------
// K' = 64 (2 tiles of 32); N = 6 padded to 16. src cw1[(kh*5+kw)*6 + n].
__global__ __launch_bounds__(256) void pack_b_conv1(const float* __restrict__ src,
                                                    _Float16* __restrict__ dst) {
    int e = blockIdx.x * 256 + threadIdx.x;
    if (e >= 2 * 512) return;
    int j    = e & 15;
    int lane = (e >> 4) & 31;
    int kt   = e >> 9;
    int kp   = kt * 32 + wmma_kof(lane, j);
    int kh   = kp >> 3;
    int kw   = kp & 7;
    int n    = lane & 15;
    float v = 0.0f;
    if (kh < 5 && kw < 5 && n < 6) v = src[(size_t)(kh * 5 + kw) * 6 + n];
    dst[e] = (_Float16)v;
}

// ---- Pack cw2 with the conv2 K-permutation: k' = kh*32 + (kw*6+ci) -------
// t = k'%32 in [0,30) maps to original k = kh*30 + t ; t=30,31 are zeros.
__global__ __launch_bounds__(256) void pack_b_conv2(const float* __restrict__ src,
                                                    _Float16* __restrict__ dst) {
    int e = blockIdx.x * 256 + threadIdx.x;
    if (e >= 5 * 512) return;
    int j    = e & 15;
    int lane = (e >> 4) & 31;
    int kt   = e >> 9;                       // == kh
    int kp   = kt * 32 + wmma_kof(lane, j);
    int kh   = kp >> 5;
    int t    = kp & 31;
    int n    = lane & 15;
    float v = 0.0f;
    if (t < 30) v = src[(size_t)(kh * 30 + t) * 16 + n];
    dst[e] = (_Float16)v;
}

// ---- conv1 (5x5, 1->6) as WMMA im2col GEMM + relu + 2x2 maxpool ----------
// One block per image (8 waves). GEMM: M=576 (36 tiles), K'=64, N=16(6 used).
// K' = kh*8+kw: lane runs are contiguous image-row segments (unaligned 16B
// global loads of f16), pad positions kw=5..7 are compile-time zeros.
// x-pool pairs sit in adjacent C VGPRs of one lane (hp index = 3*m + n,
// exactly linear in conv position m); y-pool goes via LDS.
__global__ __launch_bounds__(256) void conv1_pool(const _Float16* __restrict__ img16,
                                                  const _Float16* __restrict__ bprep,
                                                  const float* __restrict__ cb1,
                                                  _Float16* __restrict__ pooled1) {
    __shared__ float hp[24 * 12 * 6];          // x-pooled relu'd conv1 output
    int image = blockIdx.x;
    int tid   = threadIdx.x;
    int wv    = tid >> 5, lane = tid & 31;
    int half  = lane >> 4, n = lane & 15;
    const _Float16* pimg = img16 + (size_t)image * 784;
    float bv = (n < 6) ? cb1[n] : 0.0f;
    v16h b0 = *(const v16h*)(bprep + (lane << 4));           // k-tile 0
    v16h b1 = *(const v16h*)(bprep + 512 + (lane << 4));     // k-tile 1 (kh=4)
    for (int t = wv; t < 36; t += 8) {
        int mA = t * 16 + (lane & 15);         // A row = conv output position
        int oy = mA / 24, ox = mA % 24;
        const _Float16* rp = pimg + oy * 28 + ox;
        // k-tile 0: runs are rows kh = half and kh = 2+half, kw = 0..7 (5..7 pad)
        v8h r0 = *(const v8h_u*)(rp + half * 28);
        v8h r1 = *(const v8h_u*)(rp + (2 + half) * 28);
        v8h r2 = *(const v8h_u*)(rp + 4 * 28);               // kh=4 (half 0 only)
        v16h a0 = {};
        #pragma unroll
        for (int i = 0; i < 5; i++) { a0[i] = r0[i]; a0[8 + i] = r1[i]; }
        v16h a1 = {};
        #pragma unroll
        for (int i = 0; i < 5; i++) { a1[i] = half ? (_Float16)0 : r2[i]; }
        v8f c = {};
        c = __builtin_amdgcn_wmma_f32_16x16x32_f16(false, a0, false, b0,
                                                   (short)0, c, false, false);
        c = __builtin_amdgcn_wmma_f32_16x16x32_f16(false, a1, false, b1,
                                                   (short)0, c, false, false);
        if (n < 6) {
            // C row m = t*16 + 2i + 8*half ; hp flat index = 3*m + n (linear!)
            int hbase = 3 * (t * 16 + 8 * half) + n;
            #pragma unroll
            for (int i = 0; i < 4; i++) {      // x-pool: adjacent VGPR pair
                float v0 = fmaxf(c[2 * i] + bv, 0.0f);
                float v1 = fmaxf(c[2 * i + 1] + bv, 0.0f);
                hp[hbase + 6 * i] = fmaxf(v0, v1);
            }
        }
    }
    __syncthreads();
    for (int idx = tid; idx < 864; idx += 256) {             // y-pool + store
        int py = idx / 72;                     // out idx = py*72 + r
        int r  = idx - py * 72;                // r = px*6 + ch
        float v = fmaxf(hp[py * 144 + r], hp[py * 144 + 72 + r]);
        pooled1[(size_t)image * 864 + idx] = (_Float16)v;
    }
}

// ---- conv2 (5x5, 6->16) as WMMA im2col GEMM, fused relu + maxpool --------
// One wave per image: M=64 (4 tiles of 16 positions), K'=160 (5 steps of 32).
// K permuted as k' = kh*32 + (kw*6+ci): each lane's two 8-element A runs are
// contiguous f16 in pooled1 -> two unaligned 16B loads, zero VALU decode.
// C tile t: lane(half h) VGPR r holds conv out (oy=2t+h, ox=r, ch=lane&15).
__global__ __launch_bounds__(256) void conv2_pool(const _Float16* __restrict__ pooled1,
                                                  const _Float16* __restrict__ bprep,
                                                  const float* __restrict__ cb2,
                                                  _Float16* __restrict__ X) {
    int wave = (blockIdx.x * 256 + threadIdx.x) >> 5;
    int lane = threadIdx.x & 31;
    int half = lane >> 4;
    int n    = lane & 15;
    const _Float16* pimg = pooled1 + (size_t)wave * 864;
    #pragma unroll
    for (int mt = 0; mt < 4; mt++) {
        int p  = mt * 16 + (lane & 15);      // A row = output position
        int oy = p >> 3, ox = p & 7;
        const _Float16* pbase = pimg + oy * 72 + ox * 6;
        v8f acc0 = {}, acc1 = {};            // dual accumulators break RAW chain
        #pragma unroll
        for (int kh = 0; kh < 5; kh++) {
            const _Float16* rp = pbase + kh * 72;
            v8h a0 = *(const v8h_u*)(rp + half * 8);        // t = half*8 .. +7
            v8h a1 = *(const v8h_u*)(rp + 16 + half * 8);   // t = 16+half*8 .. +7
            // t = 30,31 (only reachable when half==1, elems 6,7) are K-pad -> 0
            a1[6] = half ? (_Float16)0 : a1[6];
            a1[7] = half ? (_Float16)0 : a1[7];
            v16h a = __builtin_shufflevector(a0, a1, 0, 1, 2, 3, 4, 5, 6, 7,
                                             8, 9, 10, 11, 12, 13, 14, 15);
            v16h b = *(const v16h*)(bprep + ((size_t)kh << 9) + (lane << 4));
            if (kh & 1)
                acc1 = __builtin_amdgcn_wmma_f32_16x16x32_f16(false, a, false, b,
                                                              (short)0, acc1, false, false);
            else
                acc0 = __builtin_amdgcn_wmma_f32_16x16x32_f16(false, a, false, b,
                                                              (short)0, acc0, false, false);
        }
        v8f c = acc0 + acc1;
        float bv = cb2[n];
        #pragma unroll
        for (int px = 0; px < 4; px++) {
            float v0 = fmaxf(c[2 * px] + bv, 0.0f);
            float v1 = fmaxf(c[2 * px + 1] + bv, 0.0f);
            float q  = fmaxf(v0, v1);
            // pool across oy parity: ds_swizzle SWAPX16 (xor 0x10, and 0x1f)
            float o  = __int_as_float(__builtin_amdgcn_ds_swizzle(__float_as_int(q), 0x401F));
            q = fmaxf(q, o);
            // both halves hold identical q -> duplicate same-value store is benign
            X[(size_t)wave * 256 + ((mt * 4 + px) << 4) + n] = (_Float16)q;
        }
    }
}

// ---- generic dense WMMA GEMM: one wave per 16x16 output tile -------------
template <int K, int NPAD, int BIASN, bool RELU, bool F16OUT>
__global__ __launch_bounds__(256) void dense_gemm(const _Float16* __restrict__ A,
                                                  const _Float16* __restrict__ B,
                                                  const float* __restrict__ bias,
                                                  _Float16* __restrict__ outH,
                                                  float* __restrict__ outF) {
    constexpr int TILESN = NPAD / 16;
    int wave = (blockIdx.x * 256 + threadIdx.x) >> 5;
    int lane = threadIdx.x & 31;
    int tM = wave / TILESN, tN = wave % TILESN;
    int half = lane >> 4;
    int m0   = lane & 15;
    const _Float16* arow = A + (size_t)(tM * 16 + m0) * K;
    v8f acc0 = {}, acc1 = {};                // dual accumulators break RAW chain
    #pragma unroll
    for (int k0 = 0; k0 < K; k0 += 32) {
        if (k0 + 32 < K)                       // gfx1250 global_prefetch_b8
            __builtin_prefetch((const void*)(arow + k0 + 32), 0, 1);
        v8h a0 = *(const v8h*)(arow + k0 + half * 8);
        v8h a1 = *(const v8h*)(arow + k0 + 16 + half * 8);
        v16h a = __builtin_shufflevector(a0, a1, 0, 1, 2, 3, 4, 5, 6, 7,
                                         8, 9, 10, 11, 12, 13, 14, 15);
        constexpr int KT_STRIDE = TILESN << 9;
        v16h b = *(const v16h*)(B + (size_t)(k0 >> 5) * KT_STRIDE +
                                ((size_t)tN << 9) + (lane << 4));
        if ((k0 >> 5) & 1)
            acc1 = __builtin_amdgcn_wmma_f32_16x16x32_f16(false, a, false, b,
                                                          (short)0, acc1, false, false);
        else
            acc0 = __builtin_amdgcn_wmma_f32_16x16x32_f16(false, a, false, b,
                                                          (short)0, acc0, false, false);
    }
    v8f c = acc0 + acc1;
    int n = tN * 16 + m0;
    float bv = (n < BIASN) ? bias[n] : 0.0f;
    #pragma unroll
    for (int r = 0; r < 8; r++) {
        float v = c[r] + bv;
        if (RELU) v = fmaxf(v, 0.0f);
        int m = tM * 16 + r + 8 * half;
        if (F16OUT) outH[(size_t)m * NPAD + n] = (_Float16)v;
        else        outF[(size_t)m * NPAD + n] = v;
    }
}

// ---- softmax + Luhn pmf chain: one thread per batch element --------------
__device__ __constant__ int kLuhnInv[10] = {0, 5, 1, 6, 2, 7, 3, 8, 4, 9};

__device__ __forceinline__ void softmax10(const float* r, float* p) {
    float mx = r[0];
    for (int j = 1; j < 10; j++) mx = fmaxf(mx, r[j]);
    float s = 0.0f;
    for (int j = 0; j < 10; j++) { p[j] = expf(r[j] - mx); s += p[j]; }
    float inv = 1.0f / s;
    for (int j = 0; j < 10; j++) p[j] *= inv;
}

__global__ __launch_bounds__(256) void luhn_kernel(const float* __restrict__ logits,
                                                   float* __restrict__ out) {
    int b = blockIdx.x * 256 + threadIdx.x;
    if (b >= 1024) return;
    const float* L = logits + (size_t)b * 16 * 16;   // 16 rows, ld=16
    float cd[10], check[10], t[10], dd[10], nc[10];
    softmax10(L, cd);                                // check digit (pos 0)
    for (int j = 0; j < 10; j++) check[j] = 0.0f;
    check[0] = 1.0f;
    for (int d = 1; d < 16; d++) {
        softmax10(L + d * 16, t);
        int i = d - 1;                               // m=15 -> permute odd i
        if (i & 1) { for (int j = 0; j < 10; j++) dd[j] = t[kLuhnInv[j]]; }
        else       { for (int j = 0; j < 10; j++) dd[j] = t[j]; }
        for (int s = 0; s < 10; s++) nc[s] = 0.0f;
        for (int a = 0; a < 10; a++) {
            float ca = check[a];
            for (int j = 0; j < 10; j++) {
                int s = a + j; if (s >= 10) s -= 10; // conv then mod10 fold
                nc[s] += ca * dd[j];
            }
        }
        for (int s = 0; s < 10; s++) check[s] = nc[s];
    }
    float tot = 0.0f;                                // total[10] of final conv
    for (int i = 1; i < 10; i++) tot += cd[i] * check[10 - i];
    out[b] = logf(tot);
}

extern "C" void kernel_launch(void* const* d_in, const int* in_sizes, int n_in,
                              void* d_out, int out_size, void* d_ws, size_t ws_size,
                              hipStream_t stream) {
    (void)in_sizes; (void)n_in; (void)out_size; (void)ws_size;
    const float* inputs = (const float*)d_in[0];
    const float* cw1 = (const float*)d_in[1];
    const float* cb1 = (const float*)d_in[2];
    const float* cw2 = (const float*)d_in[3];
    const float* cb2 = (const float*)d_in[4];
    const float* dw1 = (const float*)d_in[5];
    const float* db1 = (const float*)d_in[6];
    const float* dw2 = (const float*)d_in[7];
    const float* db2 = (const float*)d_in[8];
    const float* dw3 = (const float*)d_in[9];
    const float* db3 = (const float*)d_in[10];

    const int NIMG = 16384;
    const int NPIX = NIMG * 784;                     // 12,845,056

    char* w = (char*)d_ws;
    size_t o = 0;
    _Float16* IMG16   = (_Float16*)(w + o); o += (size_t)NPIX * 2 + 256;   // 25.7 MB (+pad)
    _Float16* pooled1 = (_Float16*)(w + o); o += 16384ull * 864 * 2;       // 28.3 MB
    _Float16* X       = (_Float16*)(w + o); o += 16384ull * 256 * 2;       //  8.4 MB
    _Float16* H1      = (_Float16*)(w + o); o += 16384ull * 128 * 2;       //  4.2 MB
    _Float16* H2      = (_Float16*)(w + o); o += 16384ull *  96 * 2;       //  3.1 MB
    float*    logitsB = (float*)   (w + o); o += 16384ull *  16 * 4;       //  1.0 MB
    _Float16* Bc1     = (_Float16*)(w + o); o += 2ull  * 512 * 2;          // 64x16 (perm K)
    _Float16* Bc2     = (_Float16*)(w + o); o += 5ull  * 512 * 2;          // 160x16 (perm K)
    _Float16* B1      = (_Float16*)(w + o); o += 64ull * 512 * 2;          // 256x128
    _Float16* B2      = (_Float16*)(w + o); o += 24ull * 512 * 2;          // 128x96
    _Float16* B3      = (_Float16*)(w + o); o += 3ull  * 512 * 2;          // 96x16

    // Stage weights into WMMA-ready f16 B tiles (zero-padded K/N).
    pack_b_conv1<<<4,  256, 0, stream>>>(cw1, Bc1);
    pack_b_conv2<<<10, 256, 0, stream>>>(cw2, Bc2);
    pack_b<<<128, 256, 0, stream>>>(dw1, B1, 256, 120, 8, 64 * 512);
    pack_b<<<48,  256, 0, stream>>>(dw2, B2, 120, 84, 6, 24 * 512);
    pack_b<<<6,   256, 0, stream>>>(dw3, B3, 84, 10, 1, 3 * 512);

    // f32 -> f16 image convert (16 zero-pad entries at the tail).
    cvt_f16<<<(NPIX + 16 + 255) / 256, 256, 0, stream>>>(inputs, IMG16, NPIX, NPIX + 16);

    conv1_pool<<<16384, 256, 0, stream>>>(IMG16, Bc1, cb1, pooled1);
    conv2_pool<<<2048,  256, 0, stream>>>(pooled1, Bc2, cb2, X);     // 16384 waves

    // FC1: 16384x256 @ 256x128 ; FC2: 16384x128 @ 128x96 ; FC3: -> 16384x16 f32
    dense_gemm<256, 128, 120, true,  true ><<<1024, 256, 0, stream>>>(X,  B1, db1, H1, nullptr);
    dense_gemm<128,  96,  84, true,  true ><<<768,  256, 0, stream>>>(H1, B2, db2, H2, nullptr);
    dense_gemm< 96,  16,  10, false, false><<<128,  256, 0, stream>>>(H2, B3, db3, nullptr, logitsB);

    luhn_kernel<<<4, 256, 0, stream>>>(logitsB, (float*)d_out);
}
